// PerseidByteModel_12884901888070
// MI455X (gfx1250) — matrix operand order repaired and checked by reading, not compiled, not run on hardware
//
#include <hip/hip_runtime.h>
#include <math.h>

// ---------------- model constants ----------------
#define B_   2
#define S_   1024
#define NH_  16
#define NKV_ 4
#define HD_  128
#define E_   1024
#define FF_  4096
#define L_   6
#define SW_  256
#define V_   256
#define T_   (B_ * S_)          // 2048 tokens
#define EPS_ 1e-6f
#define SCALING_ 0.0625f        // 256^-0.5

typedef __attribute__((ext_vector_type(16))) __bf16 v16bf;
typedef __attribute__((ext_vector_type(8)))  float  v8f;

union Frag16 { v16bf v; uint4 q[2]; };
union FragC  { v8f   v; float f[8]; };

__device__ __forceinline__ unsigned short f2bf(float f) {
  unsigned u = __float_as_uint(f);
  u = u + 0x7fffu + ((u >> 16) & 1u);       // RNE
  return (unsigned short)(u >> 16);
}
__device__ __forceinline__ float bf2f(unsigned short s) {
  return __uint_as_float(((unsigned)s) << 16);
}
__device__ __forceinline__ float gelu_tanh(float x) {
  float t = tanhf(0.7978845608028654f * (x + 0.044715f * x * x * x));
  return 0.5f * x * (1.0f + t);
}

// ---------------- embedding ----------------
__global__ __launch_bounds__(256) void embed_kernel(const int* __restrict__ ids,
                                                    const float* __restrict__ emb,
                                                    float* __restrict__ x) {
  int t = blockIdx.x, tid = threadIdx.x;
  const float* src = emb + (size_t)ids[t] * E_;
  float* dst = x + (size_t)t * E_;
  #pragma unroll
  for (int i = 0; i < E_ / 256; ++i) dst[tid + i * 256] = src[tid + i * 256] * 32.0f; // sqrt(E)
}

// ---------------- rope tables ----------------
__global__ __launch_bounds__(128) void rope_table_kernel(float* __restrict__ cosT,
                                                         float* __restrict__ sinT,
                                                         float base) {
  int pos = blockIdx.x, d = threadIdx.x;
  int i = d & 63;
  float freq = powf(base, -((float)i) / 64.0f);
  float ang = (float)pos * freq;
  cosT[pos * HD_ + d] = cosf(ang);
  sinT[pos * HD_ + d] = sinf(ang);
}

// ---------------- rmsnorm -> bf16 ----------------
__global__ __launch_bounds__(256) void rmsnorm_bf16_kernel(const float* __restrict__ x,
                                                           const float* __restrict__ scale,
                                                           unsigned short* __restrict__ out) {
  __shared__ float red[8];
  __shared__ float sinv;
  int row = blockIdx.x, tid = threadIdx.x;
  const float* xr = x + (size_t)row * E_;
  float s = 0.f;
  #pragma unroll
  for (int i = 0; i < E_ / 256; ++i) { float v = xr[tid + i * 256]; s += v * v; }
  for (int off = 16; off; off >>= 1) s += __shfl_xor(s, off, 32);
  if ((tid & 31) == 0) red[tid >> 5] = s;
  __syncthreads();
  if (tid == 0) { float a = 0.f; for (int i = 0; i < 8; ++i) a += red[i];
                  sinv = rsqrtf(a / (float)E_ + EPS_); }
  __syncthreads();
  float inv = sinv;
  unsigned short* o = out + (size_t)row * E_;
  #pragma unroll
  for (int i = 0; i < E_ / 256; ++i) { int e = tid + i * 256;
    o[e] = f2bf(xr[e] * inv * (1.0f + scale[e])); }
}

// ---------------- x += rmsnorm(t, scale) ----------------
__global__ __launch_bounds__(256) void resid_rms_kernel(float* __restrict__ x,
                                                        const float* __restrict__ t,
                                                        const float* __restrict__ scale) {
  __shared__ float red[8];
  __shared__ float sinv;
  int row = blockIdx.x, tid = threadIdx.x;
  const float* tr = t + (size_t)row * E_;
  float* xr = x + (size_t)row * E_;
  float s = 0.f;
  #pragma unroll
  for (int i = 0; i < E_ / 256; ++i) { float v = tr[tid + i * 256]; s += v * v; }
  for (int off = 16; off; off >>= 1) s += __shfl_xor(s, off, 32);
  if ((tid & 31) == 0) red[tid >> 5] = s;
  __syncthreads();
  if (tid == 0) { float a = 0.f; for (int i = 0; i < 8; ++i) a += red[i];
                  sinv = rsqrtf(a / (float)E_ + EPS_); }
  __syncthreads();
  #pragma unroll
  for (int i = 0; i < E_ / 256; ++i) { int e = tid + i * 256;
    xr[e] += tr[e] * sinv * (1.0f + scale[e]); }
}

// ---------------- per-head rmsnorm + rope -> bf16 ----------------
// in/out layout: [T][nh*HD]; grid (T, nh), 128 threads
__global__ __launch_bounds__(128) void qknorm_rope_kernel(const float* __restrict__ in,
                                                          const float* __restrict__ scale,
                                                          const float* __restrict__ cosT,
                                                          const float* __restrict__ sinT,
                                                          unsigned short* __restrict__ out,
                                                          int nh, float oscale) {
  __shared__ float red[4];
  __shared__ float sinv;
  __shared__ float nbuf[HD_];
  int t = blockIdx.x, hh = blockIdx.y, d = threadIdx.x;
  int s = t & (S_ - 1);
  size_t base = (size_t)t * nh * HD_ + (size_t)hh * HD_;
  float v = in[base + d];
  float sq = v * v;
  for (int off = 16; off; off >>= 1) sq += __shfl_xor(sq, off, 32);
  if ((d & 31) == 0) red[d >> 5] = sq;
  __syncthreads();
  if (d == 0) sinv = rsqrtf((red[0] + red[1] + red[2] + red[3]) / (float)HD_ + EPS_);
  __syncthreads();
  float nv = v * sinv * (1.0f + scale[d]);
  nbuf[d] = nv;
  __syncthreads();
  float rot = (d < 64) ? -nbuf[d + 64] : nbuf[d - 64];
  float o = nv * cosT[s * HD_ + d] + rot * sinT[s * HD_ + d];
  out[base + d] = f2bf(o * oscale);
}

// ---------------- bf16 WMMA GEMM ----------------
// C[M,N] = A[M,K](bf16,row-major) * W[K,N](f32,row-major; converted to bf16 on load)
// modes: 0=store f32; 1=store bf16; 2=store bf16 gelu; 3=store bf16 (d * aux[idx])
__global__ __launch_bounds__(256) void gemm_bf16_kernel(const unsigned short* __restrict__ A,
                                                        const float* __restrict__ W,
                                                        float* __restrict__ outF,
                                                        unsigned short* __restrict__ outB,
                                                        const unsigned short* __restrict__ aux,
                                                        int M, int N, int K, int mode) {
  __shared__ __align__(16) unsigned short a_lds[128 * 40]; // [row][k], pitch 40
  __shared__ __align__(16) unsigned short b_lds[128 * 40]; // [n][k] (transposed W), pitch 40
  int tid = threadIdx.x;
  int w = tid >> 5, lane = tid & 31;
  int np = lane & 15, half = lane >> 4;
  int waveM = w >> 2, waveN = w & 3;
  int n0 = blockIdx.x * 128, m0 = blockIdx.y * 128;
  FragC acc[4][2];
  #pragma unroll
  for (int mt = 0; mt < 4; ++mt)
    #pragma unroll
    for (int nt = 0; nt < 2; ++nt)
      #pragma unroll
      for (int r = 0; r < 8; ++r) acc[mt][nt].f[r] = 0.f;

  for (int k0 = 0; k0 < K; k0 += 32) {
    // stage A tile 128x32 bf16
    #pragma unroll
    for (int pass = 0; pass < 2; ++pass) {
      int c = pass * 256 + tid;
      int row = c >> 2, col = (c & 3) << 3;
      uint4 d = *(const uint4*)(A + (size_t)(m0 + row) * K + k0 + col);
      *(uint4*)(&a_lds[row * 40 + col]) = d;
    }
    // stage W tile 32x128 f32 -> transposed bf16 [n][k]
    #pragma unroll
    for (int pass = 0; pass < 4; ++pass) {
      int c = pass * 256 + tid;
      int kr = c >> 5, nc = (c & 31) << 2;
      float4 fv = *(const float4*)(W + (size_t)(k0 + kr) * N + n0 + nc);
      b_lds[(nc + 0) * 40 + kr] = f2bf(fv.x);
      b_lds[(nc + 1) * 40 + kr] = f2bf(fv.y);
      b_lds[(nc + 2) * 40 + kr] = f2bf(fv.z);
      b_lds[(nc + 3) * 40 + kr] = f2bf(fv.w);
    }
    __syncthreads();
    Frag16 af[4], bfr[2];
    #pragma unroll
    for (int mt = 0; mt < 4; ++mt) {
      int rl = waveM * 64 + mt * 16 + np;
      // A layout: half0 holds K 0-7 & 16-23, half1 holds K 8-15 & 24-31
      af[mt].q[0] = *(const uint4*)(&a_lds[rl * 40 + half * 8]);
      af[mt].q[1] = *(const uint4*)(&a_lds[rl * 40 + 16 + half * 8]);
    }
    #pragma unroll
    for (int nt = 0; nt < 2; ++nt) {
      int cl = waveN * 32 + nt * 16 + np;
      // B layout: half0 holds K 0-7 & 16-23, half1 holds K 8-15 & 24-31
      bfr[nt].q[0] = *(const uint4*)(&b_lds[cl * 40 + half * 8]);
      bfr[nt].q[1] = *(const uint4*)(&b_lds[cl * 40 + 16 + half * 8]);
    }
    #pragma unroll
    for (int mt = 0; mt < 4; ++mt)
      #pragma unroll
      for (int nt = 0; nt < 2; ++nt)
        acc[mt][nt].v = __builtin_amdgcn_wmma_f32_16x16x32_bf16(
            false, af[mt].v, false, bfr[nt].v, (short)0, acc[mt][nt].v, false, false);
    __syncthreads();
  }

  #pragma unroll
  for (int mt = 0; mt < 4; ++mt)
    #pragma unroll
    for (int nt = 0; nt < 2; ++nt)
      #pragma unroll
      for (int r = 0; r < 8; ++r) {
        int row = m0 + waveM * 64 + mt * 16 + r + 8 * half;
        int col = n0 + waveN * 32 + nt * 16 + np;
        size_t idx = (size_t)row * N + col;
        float d = acc[mt][nt].f[r];
        if (mode == 0)      outF[idx] = d;
        else if (mode == 1) outB[idx] = f2bf(d);
        else if (mode == 2) outB[idx] = f2bf(gelu_tanh(d));
        else                outB[idx] = f2bf(d * bf2f(aux[idx]));
      }
}

// ---------------- flash attention (causal, optional sliding window, GQA 4:1) ----------------
// q: [T][NH*HD] bf16 (pre-scaled), k/v: [T][NKV*HD] bf16, ctx out: [T][NH*HD] bf16
// grid (S/128, NH, B), 256 threads = 8 waves; each wave owns 16 q rows.
__global__ __launch_bounds__(256) void attention_kernel(const unsigned short* __restrict__ qbf,
                                                        const unsigned short* __restrict__ kbf,
                                                        const unsigned short* __restrict__ vbf,
                                                        unsigned short* __restrict__ ctxbf,
                                                        int is_local) {
  __shared__ __align__(16) unsigned short k_lds[32 * 136];   // [krow][d], pitch 136
  __shared__ __align__(16) unsigned short vt_lds[128 * 40];  // [d][krow], pitch 40
  __shared__ __align__(16) unsigned short p_lds[8 * 16 * 40]; // per-wave 16x32, pitch 40
  int tid = threadIdx.x;
  int w = tid >> 5, lane = tid & 31;
  int np = lane & 15, half = lane >> 4;
  int q0 = blockIdx.x * 128;
  int h = blockIdx.y, b = blockIdx.z;
  int kv = h >> 2;

  // Q fragments (held in registers for the whole kernel)
  int qrow = q0 + w * 16 + np;
  size_t qbase = ((size_t)(b * S_ + qrow)) * (NH_ * HD_) + (size_t)h * HD_;
  Frag16 qa[4];
  #pragma unroll
  for (int c = 0; c < 4; ++c) {
    qa[c].q[0] = *(const uint4*)(qbf + qbase + c * 32 + half * 8);
    qa[c].q[1] = *(const uint4*)(qbf + qbase + c * 32 + 16 + half * 8);
  }

  int kb_end = q0 / 32 + 4;   // exclusive; block-uniform
  int kb_start = 0;
  if (is_local) { int jm = q0 - (SW_ - 1); if (jm > 0) kb_start = jm >> 5; }

  float m_st[8], l_st[8];
  FragC acc[8];
  #pragma unroll
  for (int r = 0; r < 8; ++r) { m_st[r] = -1e30f; l_st[r] = 0.f; }
  #pragma unroll
  for (int t2 = 0; t2 < 8; ++t2)
    #pragma unroll
    for (int r = 0; r < 8; ++r) acc[t2].f[r] = 0.f;

  for (int kb = kb_start; kb < kb_end; ++kb) {
    int ks0 = kb * 32;
    // cooperative stage: K rows + V transposed
    #pragma unroll
    for (int pass = 0; pass < 2; ++pass) {
      int c = pass * 256 + tid;
      int kr = c >> 4, col = (c & 15) << 3;
      size_t g = ((size_t)(b * S_ + ks0 + kr)) * (NKV_ * HD_) + (size_t)kv * HD_ + col;
      uint4 kd = *(const uint4*)(kbf + g);
      *(uint4*)(&k_lds[kr * 136 + col]) = kd;
      uint4 vd = *(const uint4*)(vbf + g);
      const unsigned short* vs = (const unsigned short*)&vd;
      #pragma unroll
      for (int i2 = 0; i2 < 8; ++i2) vt_lds[(col + i2) * 40 + kr] = vs[i2];
    }
    __syncthreads();

    // scores: S(16x32) = Q(16x128) K^T, two 16x16 N-tiles, K chunks of 32
    FragC sacc[2];
    #pragma unroll
    for (int nt = 0; nt < 2; ++nt)
      #pragma unroll
      for (int r = 0; r < 8; ++r) sacc[nt].f[r] = 0.f;
    #pragma unroll
    for (int nt = 0; nt < 2; ++nt) {
      int krow = nt * 16 + np;
      #pragma unroll
      for (int c2 = 0; c2 < 4; ++c2) {
        Frag16 kf;
        kf.q[0] = *(const uint4*)(&k_lds[krow * 136 + c2 * 32 + half * 8]);
        kf.q[1] = *(const uint4*)(&k_lds[krow * 136 + c2 * 32 + 16 + half * 8]);
        sacc[nt].v = __builtin_amdgcn_wmma_f32_16x16x32_bf16(
            false, qa[c2].v, false, kf.v, (short)0, sacc[nt].v, false, false);
      }
    }

    // online softmax per row (row = r + 8*half), columns striped over 16 lanes
    unsigned short* pw = &p_lds[w * 640];
    #pragma unroll
    for (int r = 0; r < 8; ++r) {
      int i = q0 + w * 16 + r + 8 * half;
      int j0 = ks0 + np, j1 = ks0 + 16 + np;
      bool v0 = (j0 <= i) && (!is_local || (i - j0) < SW_);
      bool v1 = (j1 <= i) && (!is_local || (i - j1) < SW_);
      float s0 = v0 ? sacc[0].f[r] : -1e30f;
      float s1 = v1 ? sacc[1].f[r] : -1e30f;
      float mx = fmaxf(s0, s1);
      #pragma unroll
      for (int off = 1; off < 16; off <<= 1) mx = fmaxf(mx, __shfl_xor(mx, off, 16));
      float mnew = fmaxf(m_st[r], mx);
      float fsc = __expf(m_st[r] - mnew);
      float p0 = v0 ? __expf(s0 - mnew) : 0.f;
      float p1 = v1 ? __expf(s1 - mnew) : 0.f;
      float rs = p0 + p1;
      #pragma unroll
      for (int off = 1; off < 16; off <<= 1) rs += __shfl_xor(rs, off, 16);
      l_st[r] = l_st[r] * fsc + rs;
      m_st[r] = mnew;
      #pragma unroll
      for (int t2 = 0; t2 < 8; ++t2) acc[t2].f[r] *= fsc;
      int m = r + 8 * half;
      pw[m * 40 + np] = f2bf(p0);
      pw[m * 40 + 16 + np] = f2bf(p1);
    }
    // wave-private LDS round-trip: split dscnt wait is enough (wave32 lockstep)
    asm volatile("s_wait_dscnt 0" ::: "memory");
    Frag16 pa;
    pa.q[0] = *(const uint4*)(&pw[np * 40 + half * 8]);
    pa.q[1] = *(const uint4*)(&pw[np * 40 + 16 + half * 8]);

    // ctx += P(16x32) @ V(32x128): 8 N-tiles over d
    #pragma unroll
    for (int t2 = 0; t2 < 8; ++t2) {
      Frag16 vb;
      int d = t2 * 16 + np;
      vb.q[0] = *(const uint4*)(&vt_lds[d * 40 + half * 8]);
      vb.q[1] = *(const uint4*)(&vt_lds[d * 40 + 16 + half * 8]);
      acc[t2].v = __builtin_amdgcn_wmma_f32_16x16x32_bf16(
          false, pa.v, false, vb.v, (short)0, acc[t2].v, false, false);
    }
    __syncthreads();
  }

  #pragma unroll
  for (int t2 = 0; t2 < 8; ++t2)
    #pragma unroll
    for (int r = 0; r < 8; ++r) {
      int i = q0 + w * 16 + r + 8 * half;
      int d = t2 * 16 + np;
      float val = acc[t2].f[r] / l_st[r];
      ctxbf[((size_t)(b * S_ + i)) * (NH_ * HD_) + (size_t)h * HD_ + d] = f2bf(val);
    }
}

// ---------------- host orchestration ----------------
extern "C" void kernel_launch(void* const* d_in, const int* in_sizes, int n_in,
                              void* d_out, int out_size, void* d_ws, size_t ws_size,
                              hipStream_t stream) {
  (void)in_sizes; (void)n_in; (void)out_size; (void)ws_size;
  const int*   ids          = (const int*)d_in[0];
  const float* tok_emb      = (const float*)d_in[1];
  const float* Wq           = (const float*)d_in[2];
  const float* Wk           = (const float*)d_in[3];
  const float* Wv           = (const float*)d_in[4];
  const float* Wo           = (const float*)d_in[5];
  const float* q_scale      = (const float*)d_in[6];
  const float* k_scale      = (const float*)d_in[7];
  const float* fc1          = (const float*)d_in[8];
  const float* fc2          = (const float*)d_in[9];
  const float* fc3          = (const float*)d_in[10];
  const float* ln_in        = (const float*)d_in[11];
  const float* ln_post_attn = (const float*)d_in[12];
  const float* ln_pre_ff    = (const float*)d_in[13];
  const float* ln_post_ff   = (const float*)d_in[14];
  const float* final_scale  = (const float*)d_in[15];
  const float* out_head     = (const float*)d_in[16];
  float* out = (float*)d_out;

  // workspace carve (~82 MB)
  char* p = (char*)d_ws;
  auto carve = [&](size_t bytes) -> void* {
    void* r = (void*)p; p += (bytes + 255) & ~(size_t)255; return r;
  };
  float*          x      = (float*)carve((size_t)T_ * E_ * 4);
  unsigned short* hbf    = (unsigned short*)carve((size_t)T_ * E_ * 2);
  float*          qf     = (float*)carve((size_t)T_ * NH_ * HD_ * 4);
  float*          kfb    = (float*)carve((size_t)T_ * NKV_ * HD_ * 4);
  unsigned short* qbf    = (unsigned short*)carve((size_t)T_ * NH_ * HD_ * 2);
  unsigned short* kbf    = (unsigned short*)carve((size_t)T_ * NKV_ * HD_ * 2);
  unsigned short* vbf    = (unsigned short*)carve((size_t)T_ * NKV_ * HD_ * 2);
  unsigned short* ctxbf  = (unsigned short*)carve((size_t)T_ * NH_ * HD_ * 2);
  float*          tmp    = (float*)carve((size_t)T_ * E_ * 4);
  unsigned short* g1     = (unsigned short*)carve((size_t)T_ * FF_ * 2);
  float*          cosg   = (float*)carve((size_t)S_ * HD_ * 4);
  float*          sing   = (float*)carve((size_t)S_ * HD_ * 4);
  float*          cosl   = (float*)carve((size_t)S_ * HD_ * 4);
  float*          sinl   = (float*)carve((size_t)S_ * HD_ * 4);

  rope_table_kernel<<<dim3(S_), 128, 0, stream>>>(cosg, sing, 1000000.0f);
  rope_table_kernel<<<dim3(S_), 128, 0, stream>>>(cosl, sinl, 10000.0f);
  embed_kernel<<<dim3(T_), 256, 0, stream>>>(ids, tok_emb, x);

  const bool isLocal[L_] = {true, false, true, false, true, false};
  for (int l = 0; l < L_; ++l) {
    const float* cosT = isLocal[l] ? cosl : cosg;
    const float* sinT = isLocal[l] ? sinl : sing;

    rmsnorm_bf16_kernel<<<dim3(T_), 256, 0, stream>>>(x, ln_in + l * E_, hbf);

    gemm_bf16_kernel<<<dim3(16, 16), 256, 0, stream>>>(
        hbf, Wq + (size_t)l * E_ * (NH_ * HD_), qf, nullptr, nullptr, T_, NH_ * HD_, E_, 0);
    gemm_bf16_kernel<<<dim3(4, 16), 256, 0, stream>>>(
        hbf, Wk + (size_t)l * E_ * (NKV_ * HD_), kfb, nullptr, nullptr, T_, NKV_ * HD_, E_, 0);
    gemm_bf16_kernel<<<dim3(4, 16), 256, 0, stream>>>(
        hbf, Wv + (size_t)l * E_ * (NKV_ * HD_), nullptr, vbf, nullptr, T_, NKV_ * HD_, E_, 1);

    qknorm_rope_kernel<<<dim3(T_, NH_), 128, 0, stream>>>(
        qf, q_scale + l * HD_, cosT, sinT, qbf, NH_, SCALING_);
    qknorm_rope_kernel<<<dim3(T_, NKV_), 128, 0, stream>>>(
        kfb, k_scale + l * HD_, cosT, sinT, kbf, NKV_, 1.0f);

    attention_kernel<<<dim3(S_ / 128, NH_, B_), 256, 0, stream>>>(
        qbf, kbf, vbf, ctxbf, isLocal[l] ? 1 : 0);

    gemm_bf16_kernel<<<dim3(8, 16), 256, 0, stream>>>(
        ctxbf, Wo + (size_t)l * (NH_ * HD_) * E_, tmp, nullptr, nullptr, T_, E_, NH_ * HD_, 0);
    resid_rms_kernel<<<dim3(T_), 256, 0, stream>>>(x, tmp, ln_post_attn + l * E_);

    rmsnorm_bf16_kernel<<<dim3(T_), 256, 0, stream>>>(x, ln_pre_ff + l * E_, hbf);
    gemm_bf16_kernel<<<dim3(32, 16), 256, 0, stream>>>(
        hbf, fc1 + (size_t)l * E_ * FF_, nullptr, g1, nullptr, T_, FF_, E_, 2);     // gelu(h@fc1)
    gemm_bf16_kernel<<<dim3(32, 16), 256, 0, stream>>>(
        hbf, fc2 + (size_t)l * E_ * FF_, nullptr, g1, g1, T_, FF_, E_, 3);          // *= h@fc2
    gemm_bf16_kernel<<<dim3(8, 16), 256, 0, stream>>>(
        g1, fc3 + (size_t)l * FF_ * E_, tmp, nullptr, nullptr, T_, E_, FF_, 0);
    resid_rms_kernel<<<dim3(T_), 256, 0, stream>>>(x, tmp, ln_post_ff + l * E_);
  }

  rmsnorm_bf16_kernel<<<dim3(T_), 256, 0, stream>>>(x, final_scale, hbf);
  gemm_bf16_kernel<<<dim3(V_ / 128, 16), 256, 0, stream>>>(
      hbf, out_head, out, nullptr, nullptr, T_, V_, E_, 0);
}
